// Nystroem_28862180229289
// MI455X (gfx1250) — matrix operand order repaired and checked by reading, not compiled, not run on hardware
//
#include <hip/hip_runtime.h>
#include <stdint.h>

// ---------------------------------------------------------------------------
// Problem constants (from reference): N=131072, D=128, C=1024
// ---------------------------------------------------------------------------
constexpr int D  = 128;
constexpr int C  = 1024;
constexpr int BM = 128;                 // rows per workgroup

constexpr int XB_STRIDE   = 136;        // ushort elems per row (128 + 8 pad)
constexpr int DENS_STRIDE = 1032;       // ushort elems per row (1024 + 8 pad)
constexpr int BB_STRIDE   = 520;        // ushort elems per row (512 + 8 pad)
constexpr int DENS_BYTES  = BM * DENS_STRIDE * 2;   // 264192
constexpr int XB_BYTES    = BM * XB_STRIDE * 2;     // 34816
constexpr int BB_BYTES    = 16 * BB_STRIDE * 2;     // 16640 (two fit in XB_BYTES)
constexpr int SMEM_BYTES  = DENS_BYTES + XB_BYTES + BM * (int)sizeof(float); // 299520

typedef __attribute__((ext_vector_type(16))) __bf16        v16bf;
typedef __attribute__((ext_vector_type(8)))  float         v8f;
typedef __attribute__((ext_vector_type(8)))  unsigned int  v8u;

union FragB16 { v16bf v; v8u u; uint4 q[2]; };
union FragF32 { v8f  v; float f[8]; };

__device__ __forceinline__ unsigned short f2bf(float f) {
    unsigned u = __float_as_uint(f);
    u += 0x7FFFu + ((u >> 16) & 1u);      // round-to-nearest-even
    return (unsigned short)(u >> 16);
}
__device__ __forceinline__ unsigned pack_bf2(float a, float b) {
    return (unsigned)f2bf(a) | ((unsigned)f2bf(b) << 16);
}

// Async-stage one 16x512 bf16 half-tile of normT into a padded LDS buffer.
// 8192 elems = 1024 x b128 chunks; 256 threads x 4 chunks each.
__device__ __forceinline__ void stage_half(const unsigned short* __restrict__ normT,
                                           unsigned short* __restrict__ buf,
                                           int n0, int half, int tid) {
    #pragma unroll
    for (int j = 0; j < 4; ++j) {
        const int e  = (tid + j * 256) * 8;   // 0..8191
        const int nn = e >> 9;                // row 0..15
        const int kk = e & 511;               // col within half
        unsigned ldsoff = (unsigned)(size_t)(buf + nn * BB_STRIDE + kk);
        uint64_t ga = (uint64_t)(size_t)(normT + (size_t)(n0 + nn) * C + half * 512 + kk);
        asm volatile("global_load_async_to_lds_b128 %0, %1, off"
                     :: "v"(ldsoff), "v"(ga) : "memory");
    }
}
__device__ __forceinline__ void wait_async0() {
    asm volatile("s_wait_asynccnt 0x0" ::: "memory");
}

// ---------------------------------------------------------------------------
// Prep kernel: normT (bf16, [C][C]), centers bf16 ([C][D]), c2 = ||center||^2
// ---------------------------------------------------------------------------
__global__ void nystroem_prep(const float* __restrict__ centers,
                              const float* __restrict__ norm,
                              unsigned short* __restrict__ normT,
                              unsigned short* __restrict__ cbf,
                              float* __restrict__ c2) {
    const int b = blockIdx.x, tid = threadIdx.x;
    if (b < C) {
        for (int k = tid; k < C; k += 256)
            normT[(size_t)b * C + k] = f2bf(norm[(size_t)k * C + b]);
    } else if (b < C + 512) {
        const int i = (b - C) * 256 + tid;     // 512*256 = 131072 = C*D
        cbf[i] = f2bf(centers[i]);
    } else {
        const int c = (b - C - 512) * 256 + tid;   // 4 blocks -> 1024 centers
        const float* cr = centers + (size_t)c * D;
        float s = 0.f;
        #pragma unroll 4
        for (int k = 0; k < D; ++k) { float v = cr[k]; s += v * v; }
        c2[c] = s;
    }
}

// ---------------------------------------------------------------------------
// Fused kernel: per block of 128 rows:
//   Phase 1: dens[128][1024] = exp(-g*max(x2+c2-2*X.Ct,0))  (bf16, LDS)
//   Phase 2: out[128][1024]  = dens @ norm  (WMMA bf16, async double-buffered B)
// ---------------------------------------------------------------------------
__global__ __launch_bounds__(256, 1)
void nystroem_fused(const float* __restrict__ X,
                    const float* __restrict__ gamma,
                    const unsigned short* __restrict__ cbf,
                    const unsigned short* __restrict__ normT,
                    const float* __restrict__ c2,
                    float* __restrict__ out) {
    extern __shared__ char smem[];
    unsigned short* dens  = (unsigned short*)smem;                     // 258 KB
    unsigned short* xb    = (unsigned short*)(smem + DENS_BYTES);      // 34 KB
    unsigned short* bbuf0 = xb;                                        // ph2 half A
    unsigned short* bbuf1 = xb + BB_BYTES / 2;                         // ph2 half B
    float*          x2    = (float*)(smem + DENS_BYTES + XB_BYTES);    // 512 B

    const int tid  = threadIdx.x;
    const int lane = tid & 31;
    const int wave = tid >> 5;
    const int row0 = blockIdx.x * BM;

    if (tid < BM) x2[tid] = 0.0f;
    __syncthreads();

    // ---- stage X block as bf16 + row sum of squares (2 threads per row) ----
    {
        const int r = tid & (BM - 1);
        const int h = tid >> 7;
        const float* xrow = X + (size_t)(row0 + r) * D + h * 64;
        unsigned short* xbr = xb + r * XB_STRIDE + h * 64;
        float s = 0.f;
        #pragma unroll
        for (int k = 0; k < 64; k += 4) {
            float4 v = *(const float4*)(xrow + k);
            s += v.x * v.x + v.y * v.y + v.z * v.z + v.w * v.w;
            uint2 p;
            p.x = pack_bf2(v.x, v.y);
            p.y = pack_bf2(v.z, v.w);
            *(uint2*)(xbr + k) = p;
        }
        atomicAdd(&x2[r], s);                         // ds_add_f32
    }
    __syncthreads();

    const float gam = gamma[0];
    const int m0 = wave * 16;        // this wave's 16-row tile
    const int ml = lane & 15;
    const int hi = lane >> 4;        // lane half selects K/M sub-block

    // ---- Phase 1: densities ------------------------------------------------
    FragB16 A[4];
    #pragma unroll
    for (int kc = 0; kc < 4; ++kc) {
        const unsigned short* ap = xb + (m0 + ml) * XB_STRIDE + kc * 32 + hi * 16;
        A[kc].q[0] = *(const uint4*)ap;          // K = kc*32 + hi*16 + [0..7]
        A[kc].q[1] = *(const uint4*)(ap + 8);    // K = kc*32 + hi*16 + [8..15]
    }

    for (int c0 = 0; c0 < C; c0 += 16) {
        const int cc = c0 + ml;                  // lane's column (B: N = lane&15)
        FragF32 acc0 = {}, acc1 = {};            // split accumulators for ILP
        #pragma unroll
        for (int kc = 0; kc < 4; kc += 2) {
            FragB16 B0, B1;
            const uint4* bp0 = (const uint4*)(cbf + (size_t)cc * D + kc * 32 + hi * 16);
            const uint4* bp1 = (const uint4*)(cbf + (size_t)cc * D + (kc + 1) * 32 + hi * 16);
            B0.q[0] = bp0[0]; B0.q[1] = bp0[1];
            B1.q[0] = bp1[0]; B1.q[1] = bp1[1];
            acc0.v = __builtin_amdgcn_wmma_f32_16x16x32_bf16(
                false, A[kc].v, false, B0.v, (short)0, acc0.v, false, false);
            acc1.v = __builtin_amdgcn_wmma_f32_16x16x32_bf16(
                false, A[kc + 1].v, false, B1.v, (short)0, acc1.v, false, false);
        }
        const float c2v = c2[cc];
        #pragma unroll
        for (int v = 0; v < 8; ++v) {
            const int m = m0 + hi * 8 + v;       // C/D layout: M = half*8 + vgpr
            float cross = acc0.f[v] + acc1.f[v];
            float sqd = x2[m] + c2v - 2.0f * cross;
            sqd = fmaxf(sqd, 0.0f);
            dens[m * DENS_STRIDE + cc] = f2bf(__expf(-gam * sqd));
        }
    }
    __syncthreads();   // all waves done with xb + own dens rows complete

    // ---- Phase 2: out = dens @ norm ---------------------------------------
    // Prefetch first B half-tile while caching A in registers.
    stage_half(normT, bbuf0, 0, 0, tid);

    // Cache the wave's entire A operand (16 rows x K=1024) in registers:
    // 32 chunks x 8 VGPRs = 256 VGPRs -> hot loop is 2 ds_b128 + 1 wmma.
    FragB16 A2[32];
    #pragma unroll
    for (int kc = 0; kc < 32; ++kc) {
        const unsigned short* ap = dens + (m0 + ml) * DENS_STRIDE + kc * 32 + hi * 16;
        A2[kc].q[0] = *(const uint4*)ap;
        A2[kc].q[1] = *(const uint4*)(ap + 8);
    }

    wait_async0();
    __syncthreads();     // bbuf0 holds tile0 half0

    for (int n0 = 0; n0 < C; n0 += 16) {
        // Fill half1 of the current tile while computing on half0.
        stage_half(normT, bbuf1, n0, 1, tid);

        FragF32 acc0 = {}, acc1 = {};
        #pragma unroll
        for (int kc = 0; kc < 16; kc += 2) {          // K = 0..511 from bbuf0
            FragB16 b0, b1;
            const unsigned short* bp0 = bbuf0 + ml * BB_STRIDE + kc * 32 + hi * 16;
            const unsigned short* bp1 = bbuf0 + ml * BB_STRIDE + (kc + 1) * 32 + hi * 16;
            b0.q[0] = *(const uint4*)bp0;  b0.q[1] = *(const uint4*)(bp0 + 8);
            b1.q[0] = *(const uint4*)bp1;  b1.q[1] = *(const uint4*)(bp1 + 8);
            acc0.v = __builtin_amdgcn_wmma_f32_16x16x32_bf16(
                false, A2[kc].v, false, b0.v, (short)0, acc0.v, false, false);
            acc1.v = __builtin_amdgcn_wmma_f32_16x16x32_bf16(
                false, A2[kc + 1].v, false, b1.v, (short)0, acc1.v, false, false);
        }

        wait_async0();
        __syncthreads();   // bbuf1 ready; everyone done reading bbuf0

        // Fill half0 of the NEXT tile while computing on half1.
        const int n0n = (n0 + 16) & (C - 1);
        stage_half(normT, bbuf0, n0n, 0, tid);

        #pragma unroll
        for (int kc = 16; kc < 32; kc += 2) {         // K = 512..1023 from bbuf1
            FragB16 b0, b1;
            const unsigned short* bp0 = bbuf1 + ml * BB_STRIDE + (kc - 16) * 32 + hi * 16;
            const unsigned short* bp1 = bbuf1 + ml * BB_STRIDE + (kc - 15) * 32 + hi * 16;
            b0.q[0] = *(const uint4*)bp0;  b0.q[1] = *(const uint4*)(bp0 + 8);
            b1.q[0] = *(const uint4*)bp1;  b1.q[1] = *(const uint4*)(bp1 + 8);
            acc0.v = __builtin_amdgcn_wmma_f32_16x16x32_bf16(
                false, A2[kc].v, false, b0.v, (short)0, acc0.v, false, false);
            acc1.v = __builtin_amdgcn_wmma_f32_16x16x32_bf16(
                false, A2[kc + 1].v, false, b1.v, (short)0, acc1.v, false, false);
        }

        float* op = out + (size_t)(row0 + m0 + hi * 8) * C + n0 + ml;
        #pragma unroll
        for (int v = 0; v < 8; ++v) op[(size_t)v * C] = acc0.f[v] + acc1.f[v];

        wait_async0();
        __syncthreads();   // bbuf0 (next tile) ready; everyone done with bbuf1
    }
}

// ---------------------------------------------------------------------------
// Host entry
// ---------------------------------------------------------------------------
extern "C" void kernel_launch(void* const* d_in, const int* in_sizes, int n_in,
                              void* d_out, int out_size, void* d_ws, size_t ws_size,
                              hipStream_t stream) {
    (void)n_in; (void)out_size; (void)ws_size;
    const float* X       = (const float*)d_in[0];   // [N, D]
    const float* centers = (const float*)d_in[1];   // [C, D]
    const float* gamma   = (const float*)d_in[2];   // [1]
    const float* norm    = (const float*)d_in[3];   // [C, C]
    float* out = (float*)d_out;

    char* ws = (char*)d_ws;
    unsigned short* normT = (unsigned short*)ws;                               // 2 MB
    unsigned short* cbf   = (unsigned short*)(ws + (size_t)C * C * 2);         // 256 KB
    float*          c2    = (float*)(ws + (size_t)C * C * 2 + (size_t)C * D * 2); // 4 KB

    const int N = in_sizes[0] / D;   // 131072

    hipFuncSetAttribute((const void*)nystroem_fused,
                        hipFuncAttributeMaxDynamicSharedMemorySize, SMEM_BYTES);

    nystroem_prep<<<C + 512 + 4, 256, 0, stream>>>(centers, norm, normT, cbf, c2);
    nystroem_fused<<<N / BM, 256, SMEM_BYTES, stream>>>(X, gamma, cbf, normT, c2, out);
}
